// KVLoRA_88158498718371
// MI455X (gfx1250) — matrix operand order repaired and checked by reading, not compiled, not run on hardware
//
#include <hip/hip_runtime.h>

typedef __attribute__((ext_vector_type(16))) __bf16 v16bf;
typedef __attribute__((ext_vector_type(8)))  __bf16 v8bf;
typedef __attribute__((ext_vector_type(8)))  float  v8f;
typedef __attribute__((ext_vector_type(4)))  int    v4i;

#define D_DIM 1024
#define N_DIM 3072          // 3*D
#define M_DIM 32768         // B*S = 32*1024
#define R_DIM 5
#define BM 256              // block tile M
#define BN 128              // block tile N
#define LDS_STRIDE 72       // 64 elems + 8 pad (16B) -> conflict-free banks
#define KTILES (D_DIM / 64) // 16

#if __has_builtin(__builtin_amdgcn_global_load_async_to_lds_b128)
#define USE_ASYNC_LDS 1
#define GPTR(p) ((__attribute__((address_space(1))) v4i*)(p))
#define LPTR(p) ((__attribute__((address_space(3))) v4i*)(p))
#endif

// ---------------------------------------------------------------------------
// Kernel 1: Wb[o,d] = bf16( W[o,d] + delta[o,d] )
// ---------------------------------------------------------------------------
__global__ __launch_bounds__(256) void kvlora_prep_w(
    const float* __restrict__ W,  const float* __restrict__ A0,
    const float* __restrict__ A1, const float* __restrict__ B0,
    const float* __restrict__ B1, const float* __restrict__ s0,
    const float* __restrict__ s1, __bf16* __restrict__ Wb)
{
    int idx = blockIdx.x * 256 + threadIdx.x;   // [0, 3072*1024)
    int o = idx >> 10;
    int d = idx & 1023;
    float delta = 0.f;
    if (o >= D_DIM) {
        if (o < 2 * D_DIM) {
            int i = o - D_DIM;
            float acc = 0.f;
            #pragma unroll
            for (int r = 0; r < R_DIM; ++r)
                acc += B0[i * R_DIM + r] * A0[r * D_DIM + d];
            delta = s0[0] * acc;
        } else {
            int i = o - 2 * D_DIM;
            float acc = 0.f;
            #pragma unroll
            for (int r = 0; r < R_DIM; ++r)
                acc += B1[i * R_DIM + r] * A1[r * D_DIM + d];
            delta = s1[0] * acc;
        }
    }
    Wb[idx] = (__bf16)(W[idx] + delta);
}

// ---------------------------------------------------------------------------
// Kernel 2: Xb = bf16(x), 8 elements per thread
// ---------------------------------------------------------------------------
__global__ __launch_bounds__(256) void kvlora_cvt_x(
    const float* __restrict__ x, __bf16* __restrict__ Xb)
{
    long t    = (long)blockIdx.x * 256 + threadIdx.x;
    long base = t * 8;
    float4 f0 = ((const float4*)(x + base))[0];
    float4 f1 = ((const float4*)(x + base))[1];
    v8bf o;
    o[0] = (__bf16)f0.x; o[1] = (__bf16)f0.y;
    o[2] = (__bf16)f0.z; o[3] = (__bf16)f0.w;
    o[4] = (__bf16)f1.x; o[5] = (__bf16)f1.y;
    o[6] = (__bf16)f1.z; o[7] = (__bf16)f1.w;
    *(v8bf*)(Xb + base) = o;
}

// ---------------------------------------------------------------------------
// WMMA compute over one 256x128x64 LDS tile (per-wave 64x64 sub-tile, 4x4)
// ---------------------------------------------------------------------------
__device__ __forceinline__ void compute_tile(
    const __bf16* __restrict__ lXbuf, const __bf16* __restrict__ lWbuf,
    int waveM, int waveN, int l16, int hi, v8f acc[4][4])
{
    #pragma unroll
    for (int kk = 0; kk < 64; kk += 32) {
        // A fragments (16x32 MxK): lanes 0-15 K {kk..+7, kk+16..+23},
        // lanes 16-31 K {kk+8..+15, kk+24..+31}
        v16bf afrag[4];
        #pragma unroll
        for (int i = 0; i < 4; ++i) {
            const __bf16* src = &lXbuf[(waveM + i * 16 + l16) * LDS_STRIDE + kk + hi * 8];
            v8bf lo = *(const v8bf*)(src);
            v8bf hh = *(const v8bf*)(src + 16);
            #pragma unroll
            for (int t = 0; t < 8; ++t) { afrag[i][t] = lo[t]; afrag[i][t + 8] = hh[t]; }
        }
        // B fragments (32x16 KxN): lane = n, lanes 0-15 K {kk..+7, kk+16..+23},
        // lanes 16-31 K {kk+8..+15, kk+24..+31}
        v16bf bfrag[4];
        #pragma unroll
        for (int j = 0; j < 4; ++j) {
            const __bf16* src = &lWbuf[(waveN + j * 16 + l16) * LDS_STRIDE + kk + hi * 8];
            v8bf lo = *(const v8bf*)(src);
            v8bf hh = *(const v8bf*)(src + 16);
            #pragma unroll
            for (int t = 0; t < 8; ++t) { bfrag[j][t] = lo[t]; bfrag[j][t + 8] = hh[t]; }
        }
        #pragma unroll
        for (int i = 0; i < 4; ++i)
            #pragma unroll
            for (int j = 0; j < 4; ++j)
                acc[i][j] = __builtin_amdgcn_wmma_f32_16x16x32_bf16(
                    false, afrag[i], false, bfrag[j], (short)0, acc[i][j], false, false);
    }
}

// ---------------------------------------------------------------------------
// Kernel 3: out[m,n] = sum_k Xb[m,k]*Wb[n,k] + bias[n]
// 256x128 block tile, 8 waves (4M x 2N), 64x64 per wave.
// Double-buffered LDS fed by async global->LDS (ASYNCcnt).
// ---------------------------------------------------------------------------
__global__ __launch_bounds__(256) void kvlora_gemm(
    const __bf16* __restrict__ Xb, const __bf16* __restrict__ Wb,
    const float* __restrict__ bias, float* __restrict__ out)
{
    __shared__ __bf16 lX[2][BM * LDS_STRIDE];   // 2 x 36 KB
    __shared__ __bf16 lW[2][BN * LDS_STRIDE];   // 2 x 18 KB

    const int tid  = threadIdx.x;
    const int lane = tid & 31;
    const int wave = tid >> 5;
    const int n0   = blockIdx.x * BN;
    const int m0   = blockIdx.y * BM;

    const int waveM = (wave & 3) * 64;      // 4 waves along M
    const int waveN = (wave >> 2) * 64;     // 2 waves along N
    const int l16   = lane & 15;
    const int hi    = lane >> 4;

    v8f acc[4][4] = {};

    // loader mapping: X -> one full 64-col row per thread (8 x b128)
    //                 W -> half row per thread            (4 x b128)
    const __bf16* gX = Xb + (size_t)(m0 + tid) * D_DIM;
    const int lOffX  = tid * LDS_STRIDE;
    const __bf16* gW = Wb + (size_t)(n0 + (tid >> 1)) * D_DIM + (tid & 1) * 32;
    const int lOffW  = (tid >> 1) * LDS_STRIDE + (tid & 1) * 32;

#ifdef USE_ASYNC_LDS
    #pragma unroll
    for (int c = 0; c < 64; c += 8)
        __builtin_amdgcn_global_load_async_to_lds_b128(
            GPTR(gX + c), LPTR(&lX[0][lOffX + c]), 0, 0);
    #pragma unroll
    for (int c = 0; c < 32; c += 8)
        __builtin_amdgcn_global_load_async_to_lds_b128(
            GPTR(gW + c), LPTR(&lW[0][lOffW + c]), 0, 0);

    #pragma unroll 1
    for (int t = 0; t < KTILES; ++t) {
#if __has_builtin(__builtin_amdgcn_s_wait_asynccnt)
        __builtin_amdgcn_s_wait_asynccnt(0);
#else
        asm volatile("s_wait_asynccnt 0" ::: "memory");
#endif
        __syncthreads();                    // tile t resident for all waves
        if (t + 1 < KTILES) {
            const int kb = (t + 1) * 64;
            const int nb = (t + 1) & 1;
            #pragma unroll
            for (int c = 0; c < 64; c += 8)
                __builtin_amdgcn_global_load_async_to_lds_b128(
                    GPTR(gX + kb + c), LPTR(&lX[nb][lOffX + c]), 0, 0);
            #pragma unroll
            for (int c = 0; c < 32; c += 8)
                __builtin_amdgcn_global_load_async_to_lds_b128(
                    GPTR(gW + kb + c), LPTR(&lW[nb][lOffW + c]), 0, 0);
        }
        compute_tile(lX[t & 1], lW[t & 1], waveM, waveN, l16, hi, acc);
    }
#else
    // ---- fallback: register-staged prefetch pipeline ----
    v8bf rx[8], rw[4];
    #pragma unroll
    for (int c = 0; c < 8; ++c) rx[c] = *(const v8bf*)(gX + c * 8);
    #pragma unroll
    for (int c = 0; c < 4; ++c) rw[c] = *(const v8bf*)(gW + c * 8);
    #pragma unroll 1
    for (int t = 0; t < KTILES; ++t) {
        __bf16* bx = lX[t & 1];
        __bf16* bw = lW[t & 1];
        #pragma unroll
        for (int c = 0; c < 8; ++c) *(v8bf*)(&bx[lOffX + c * 8]) = rx[c];
        #pragma unroll
        for (int c = 0; c < 4; ++c) *(v8bf*)(&bw[lOffW + c * 8]) = rw[c];
        __syncthreads();
        if (t + 1 < KTILES) {
            const int kb = (t + 1) * 64;
            #pragma unroll
            for (int c = 0; c < 8; ++c) rx[c] = *(const v8bf*)(gX + kb + c * 8);
            #pragma unroll
            for (int c = 0; c < 4; ++c) rw[c] = *(const v8bf*)(gW + kb + c * 8);
        }
        compute_tile(bx, bw, waveM, waveN, l16, hi, acc);
    }
#endif

    // ---- epilogue: +bias, f32 store in C layout ----
    #pragma unroll
    for (int j = 0; j < 4; ++j) {
        const int n = n0 + waveN + j * 16 + l16;
        const float bv = bias[n];
        #pragma unroll
        for (int i = 0; i < 4; ++i) {
            const int mbase = m0 + waveM + i * 16 + hi * 8;
            #pragma unroll
            for (int r = 0; r < 8; ++r)
                out[(size_t)(mbase + r) * N_DIM + n] = acc[i][j][r] + bv;
        }
    }
}

// ---------------------------------------------------------------------------
extern "C" void kernel_launch(void* const* d_in, const int* in_sizes, int n_in,
                              void* d_out, int out_size, void* d_ws, size_t ws_size,
                              hipStream_t stream) {
    (void)in_sizes; (void)n_in; (void)out_size; (void)ws_size;
    const float* x    = (const float*)d_in[0];
    const float* W    = (const float*)d_in[1];
    const float* bias = (const float*)d_in[2];
    const float* A0   = (const float*)d_in[3];
    const float* A1   = (const float*)d_in[4];
    const float* B0   = (const float*)d_in[5];
    const float* B1   = (const float*)d_in[6];
    const float* s0   = (const float*)d_in[7];
    const float* s1   = (const float*)d_in[8];
    float* out = (float*)d_out;

    __bf16* Xb = (__bf16*)d_ws;                                      // 64 MB
    __bf16* Wb = (__bf16*)((char*)d_ws + (size_t)M_DIM * D_DIM * 2); // 6 MB

    kvlora_prep_w<<<(N_DIM * D_DIM) / 256, 256, 0, stream>>>(
        W, A0, A1, B0, B1, s0, s1, Wb);

    kvlora_cvt_x<<<(int)(((long)M_DIM * D_DIM / 8) / 256), 256, 0, stream>>>(x, Xb);

    dim3 grid(N_DIM / BN, M_DIM / BM);
    kvlora_gemm<<<grid, 256, 0, stream>>>(Xb, Wb, bias, out);
}
